// VQVAE_63977832841424
// MI455X (gfx1250) — matrix operand order repaired and checked by reading, not compiled, not run on hardware
//
#include <hip/hip_runtime.h>
#include <math.h>

#ifndef __has_builtin
#define __has_builtin(x) 0
#endif

// ---------------- types for WMMA ----------------
typedef __bf16 bf16x16 __attribute__((ext_vector_type(16)));
typedef float  f32x8   __attribute__((ext_vector_type(8)));

#if __has_builtin(__builtin_amdgcn_global_load_async_to_lds_b128)
#define VQVAE_ASYNC_LDS 1
// builtin signature (from hipcc diagnostic): (v4i addrspace(1)*, v4i addrspace(3)*, imm, imm)
typedef int v4i __attribute__((vector_size(4 * sizeof(int))));
typedef __attribute__((address_space(1))) v4i g_v4i;
typedef __attribute__((address_space(3))) v4i l_v4i;
#else
#define VQVAE_ASYNC_LDS 0
#endif

__device__ __forceinline__ unsigned short f2bf(float f) {
    unsigned int u = __builtin_bit_cast(unsigned int, f);
    unsigned int lsb = (u >> 16) & 1u;
    u += 0x7fffu + lsb;                       // round-to-nearest-even
    return (unsigned short)(u >> 16);
}

__device__ __forceinline__ bf16x16 mk_frag(uint4 lo, uint4 hi) {
    bf16x16 f;
    const unsigned short* pl = reinterpret_cast<const unsigned short*>(&lo);
    const unsigned short* ph = reinterpret_cast<const unsigned short*>(&hi);
    #pragma unroll
    for (int i = 0; i < 8; ++i) {
        f[i]     = __builtin_bit_cast(__bf16, pl[i]);
        f[8 + i] = __builtin_bit_cast(__bf16, ph[i]);
    }
    return f;
}

// ---------------- weight packing: fp32 -> bf16, padded [Mp x Kp] ----------------
// regular conv:  w[cout][cin][ksz] -> dst[m][ci*ksz+t] = w[m,ci,t]
// tconv:         w[cin][cout][ksz] -> dst[m][ci*ksz+t] = w[ci,m,ksz-1-t]
__global__ void vqvae_pack_weight(const float* __restrict__ w,
                                  int cout, int cin, int ksz,
                                  int Mp, int Kp, int transposed,
                                  unsigned short* __restrict__ dst)
{
    long long total = (long long)Mp * Kp;
    for (long long i = (long long)blockIdx.x * 256 + threadIdx.x; i < total;
         i += (long long)gridDim.x * 256) {
        int m  = (int)(i / Kp);
        int kk = (int)(i - (long long)m * Kp);
        float val = 0.0f;
        if (m < cout && kk < cin * ksz) {
            int ci = kk / ksz;
            int t  = kk - ci * ksz;
            val = transposed ? w[((long long)ci * cout + m) * ksz + (ksz - 1 - t)]
                             : w[((long long)m * cin + ci) * ksz + t];
        }
        dst[i] = f2bf(val);
    }
}

// ---------------- im2col: fp32 activations -> bf16 matrix, TRANSPOSED [N x Kp] ----
// dst[n*Kp + kk]: row n holds all K for that output column -> contiguous 128B rows,
// so GEMM B-tiles are plain row-segment copies (async-to-LDS friendly).
__global__ void vqvae_im2col(const float* __restrict__ src,
                             long long sC, long long sB,
                             int Lin, int cin, int ksz, int stride, int pad,
                             int Lout, int N, int Kp, int transposed,
                             unsigned short* __restrict__ dst)
{
    long long total = (long long)Kp * N;
    for (long long i = (long long)blockIdx.x * 256 + threadIdx.x; i < total;
         i += (long long)gridDim.x * 256) {
        int n  = (int)(i / Kp);
        int kk = (int)(i - (long long)n * Kp);
        float val = 0.0f;
        if (kk < cin * ksz) {
            int ci = kk / ksz;
            int t  = kk - ci * ksz;
            int b  = n / Lout;
            int l  = n - b * Lout;
            if (!transposed) {
                int pos = l * stride + t - pad;
                if (pos >= 0 && pos < Lin)
                    val = src[(long long)ci * sC + (long long)b * sB + pos];
            } else {
                int j = l + t - pad;   // lhs-dilated by 2: only even taps live
                if (j >= 0 && j <= 2 * Lin - 2 && (j & 1) == 0)
                    val = src[(long long)ci * sC + (long long)b * sB + (j >> 1)];
            }
        }
        dst[i] = f2bf(val);
    }
}

// ---------------- bf16 WMMA GEMM: out = act(A*Bt^T + bias + skip) ----------------
// A : [Mp x Kp] bf16 row-major (weights), Mp % 64 == 0, Kp % 64 == 0.
// Bt: [N  x Kp] bf16 row-major (transposed im2col), N % 128 == 0.
// 256 threads = 8 waves; block tile 64(M) x 128(N); wave tile 16(M) x 64(N):
// 4 f32x8 accumulators, K stepped by 64 -> 8 v_wmma per barrier pair.
__global__ __launch_bounds__(256) void vqvae_gemm_wmma_bf16(
    const unsigned short* __restrict__ A,
    const unsigned short* __restrict__ Bt,
    const float* __restrict__ bias,
    const float* __restrict__ skip,   // CNL [Mvalid x N] or nullptr
    float* __restrict__ out,
    int Mp, int Mvalid, int N, int Kp,
    int Lout, long long sM, long long sB, int relu)
{
    __shared__ alignas(16) unsigned short ldsA[64][64];    // [m][k]  8 KB
    __shared__ alignas(16) unsigned short ldsB[128][64];   // [n][k] 16 KB

    const int tid  = threadIdx.x;
    const int lane = tid & 31;
    const int wave = tid >> 5;
    const int mOff = (wave & 3) * 16;
    const int nOff = (wave >> 2) * 64;
    const int mBlock = blockIdx.y * 64;
    const int nBlock = blockIdx.x * 128;

    f32x8 acc[4] = {};

    const unsigned short* gA = A  + (long long)mBlock * Kp;
    const unsigned short* gB = Bt + (long long)nBlock * Kp;

    for (int k0 = 0; k0 < Kp; k0 += 64) {
#if VQVAE_ASYNC_LDS
        // ---- async global -> LDS row-segment copies (16B per lane per issue) ----
        #pragma unroll
        for (int j = 0; j < 2; ++j) {              // A tile: 64 rows x 128B
            int c = tid + j * 256;
            int r = c >> 3, cs = (c & 7) * 8;
            __builtin_amdgcn_global_load_async_to_lds_b128(
                (g_v4i*)(gA + (long long)r * Kp + k0 + cs),
                (l_v4i*)(&ldsA[r][cs]), 0, 0);
        }
        #pragma unroll
        for (int j = 0; j < 4; ++j) {              // B tile: 128 rows x 128B
            int c = tid + j * 256;
            int r = c >> 3, cs = (c & 7) * 8;
            __builtin_amdgcn_global_load_async_to_lds_b128(
                (g_v4i*)(gB + (long long)r * Kp + k0 + cs),
                (l_v4i*)(&ldsB[r][cs]), 0, 0);
        }
#if __has_builtin(__builtin_amdgcn_s_wait_asynccnt)
        __builtin_amdgcn_s_wait_asynccnt(0);
#else
        asm volatile("s_wait_asynccnt 0x0" ::: "memory");
#endif
#else
        // ---- fallback: vector loads -> b128 LDS stores ----
        uint4 va[2], vb[4];
        #pragma unroll
        for (int j = 0; j < 2; ++j) {
            int c = tid + j * 256;
            int r = c >> 3, cs = (c & 7) * 8;
            const unsigned short* p = gA + (long long)r * Kp + k0 + cs;
            va[j] = *reinterpret_cast<const uint4*>(p);
            if (k0 + 64 < Kp) __builtin_prefetch(p + 64, 0, 1);
        }
        #pragma unroll
        for (int j = 0; j < 4; ++j) {
            int c = tid + j * 256;
            int r = c >> 3, cs = (c & 7) * 8;
            vb[j] = *reinterpret_cast<const uint4*>(gB + (long long)r * Kp + k0 + cs);
        }
        #pragma unroll
        for (int j = 0; j < 2; ++j) {
            int c = tid + j * 256;
            int r = c >> 3, cs = (c & 7) * 8;
            *reinterpret_cast<uint4*>(&ldsA[r][cs]) = va[j];
        }
        #pragma unroll
        for (int j = 0; j < 4; ++j) {
            int c = tid + j * 256;
            int r = c >> 3, cs = (c & 7) * 8;
            *reinterpret_cast<uint4*>(&ldsB[r][cs]) = vb[j];
        }
#endif
        __syncthreads();

        #pragma unroll
        for (int ks = 0; ks < 64; ks += 32) {
            // A fragment (16x32): lane<16 -> K 0..7 & 16..23 ; lane>=16 -> K 8..15 & 24..31
            bf16x16 afrag;
            {
                int mRow = mOff + (lane & 15);
                int kb   = ks + ((lane >> 4) * 8);
                uint4 lo = *reinterpret_cast<const uint4*>(&ldsA[mRow][kb]);
                uint4 hi = *reinterpret_cast<const uint4*>(&ldsA[mRow][kb + 16]);
                afrag = mk_frag(lo, hi);
            }
            // 4 B fragments (32x16): lane<16 -> K 0..7 & 16..23 ; lane>=16 -> K 8..15 & 24..31
            #pragma unroll
            for (int s = 0; s < 4; ++s) {
                int nRow = nOff + s * 16 + (lane & 15);
                int kb2  = ks + ((lane >> 4) * 8);
                uint4 lo = *reinterpret_cast<const uint4*>(&ldsB[nRow][kb2]);
                uint4 hi = *reinterpret_cast<const uint4*>(&ldsB[nRow][kb2 + 16]);
                bf16x16 bfrag = mk_frag(lo, hi);
                acc[s] = __builtin_amdgcn_wmma_f32_16x16x32_bf16(
                    false, afrag, false, bfrag, (short)0, acc[s], false, false);
            }
        }
        __syncthreads();
    }

    // ---- epilogue: lanes 0-15 -> M=r, lanes 16-31 -> M=8+r; N = lane&15 ----
    const int mBase = mBlock + mOff + ((lane >> 4) * 8);
    const int nB0   = nBlock + nOff + (lane & 15);
    #pragma unroll
    for (int s = 0; s < 4; ++s) {
        int n = nB0 + s * 16;
        int b = n / Lout, l = n - b * Lout;
        #pragma unroll
        for (int r = 0; r < 8; ++r) {
            int m = mBase + r;
            if (m < Mvalid) {
                float v = acc[s][r] + (bias ? bias[m] : 0.0f);
                if (skip) v += skip[(long long)m * N + n];
                if (relu) v = fmaxf(v, 0.0f);
                out[(long long)m * sM + (long long)b * sB + l] = v;
            }
        }
    }
}

// ---------------- VQ helpers ----------------
__global__ void vqvae_cnorm(const float* __restrict__ cb, float* __restrict__ cn, int D)
{
    int j = threadIdx.x;   // 1 block x 256
    float s = 0.0f;
    for (int d = 0; d < D; ++d) {
        float v = cb[(long long)j * D + d];
        s += v * v;
    }
    cn[j] = s;
}

__global__ void vqvae_reset(int* __restrict__ hist, float* __restrict__ accum)
{
    int t = threadIdx.x;
    if (t < 256) hist[t] = 0;
    if (t == 0)  accum[0] = 0.0f;
}

__global__ void vqvae_argmin(const float* __restrict__ scores,
                             const float* __restrict__ cnorm,
                             int* __restrict__ idx, int* __restrict__ hist, int N)
{
    int n = blockIdx.x * 256 + threadIdx.x;
    if (n >= N) return;
    float best = 3.4e38f;
    int bi = 0;
    for (int j = 0; j < 256; ++j) {
        float s = cnorm[j] - 2.0f * scores[(long long)j * N + n];
        if (s < best) { best = s; bi = j; }
    }
    idx[n] = bi;
    atomicAdd(&hist[bi], 1);
}

__global__ void vqvae_quant_loss(const float* __restrict__ z,   // CNL [D x N]
                                 const float* __restrict__ cb,
                                 const int* __restrict__ idx,
                                 float* __restrict__ q,
                                 float* __restrict__ accum, int N, int D)
{
    long long total = (long long)D * N;
    float local = 0.0f;
    for (long long i = (long long)blockIdx.x * 256 + threadIdx.x; i < total;
         i += (long long)gridDim.x * 256) {
        int d = (int)(i / N);
        int n = (int)(i - (long long)d * N);
        float qv = cb[(long long)idx[n] * D + d];
        float zv = z[(long long)d * N + n];
        q[(long long)d * N + n] = qv;
        float df = qv - zv;
        local += df * df;
    }
    #pragma unroll
    for (int off = 16; off > 0; off >>= 1)
        local += __shfl_down(local, off, 32);
    if ((threadIdx.x & 31) == 0) atomicAdd(accum, local);
}

__global__ void vqvae_finalize(const int* __restrict__ hist,
                               const float* __restrict__ accum,
                               float* __restrict__ out2, float ntok, float nelem)
{
    if (threadIdx.x == 0) {
        out2[0] = 0.25f * accum[0] / nelem;          // commit loss
        float H = 0.0f;
        for (int j = 0; j < 256; ++j) {
            float p = (float)hist[j] / ntok;
            H += p * logf(p + 1e-10f);
        }
        out2[1] = expf(-H);                           // perplexity
    }
}

// ---------------- host-side orchestration ----------------
static void run_conv(hipStream_t stream,
                     const float* src, long long sC, long long sB,
                     int Lin, int cin,
                     const float* w, const float* bias, int cout,
                     int ksz, int stride, int pad, int transposed,
                     const float* skip, int relu,
                     float* out, long long osM, long long osB, int Lout,
                     unsigned short* wmat, unsigned short* bmat)
{
    const int N  = 64 * Lout;
    const int Mp = (cout + 63) & ~63;        // pad M to 64 -> unguarded tiles
    const int Kp = (cin * ksz + 63) & ~63;   // pad K to 64 -> two 32-K substeps

    {
        long long total = (long long)Mp * Kp;
        long long nb = (total + 255) / 256;
        int blocks = (int)(nb > 65536 ? 65536 : nb);
        vqvae_pack_weight<<<blocks, 256, 0, stream>>>(w, cout, cin, ksz, Mp, Kp,
                                                      transposed, wmat);
    }
    {
        long long total = (long long)Kp * N;
        long long nb = (total + 255) / 256;
        int blocks = (int)(nb > 1048576 ? 1048576 : nb);
        vqvae_im2col<<<blocks, 256, 0, stream>>>(src, sC, sB, Lin, cin, ksz, stride,
                                                 pad, Lout, N, Kp, transposed, bmat);
    }
    {
        dim3 grid(N / 128, Mp / 64);
        vqvae_gemm_wmma_bf16<<<grid, 256, 0, stream>>>(wmat, bmat, bias, skip, out,
                                                       Mp, cout, N, Kp, Lout, osM,
                                                       osB, relu);
    }
}

static void down_block(hipStream_t s, const float* src, long long sC, long long sB,
                       int Lin, int cin, int cout,
                       const float* wa, const float* ba, const float* wb,
                       const float* bb, const float* wsk, const float* bsk,
                       float* out, long long osM, long long osB,
                       float* H1, float* H2, unsigned short* wmat, unsigned short* bmat)
{
    const int Lout = Lin / 2;
    const long long N = 64LL * Lout;
    run_conv(s, src, sC, sB, Lin, cin, wa, ba, cout, 3, 2, 1, 0, nullptr, 1,
             H1, N, Lout, Lout, wmat, bmat);
    run_conv(s, H1, N, Lout, Lout, cout, wb, bb, cout, 3, 1, 1, 0, nullptr, 0,
             H2, N, Lout, Lout, wmat, bmat);
    run_conv(s, src, sC, sB, Lin, cin, wsk, bsk, cout, 1, 2, 0, 0, H2, 1,
             out, osM, osB, Lout, wmat, bmat);
}

static void up_block(hipStream_t s, const float* src, long long sC, long long sB,
                     int Lin, int cin, int cout,
                     const float* wa, const float* ba, const float* wb,
                     const float* bb, const float* wsk, const float* bsk,
                     float* out, long long osM, long long osB,
                     float* H1, float* H2, unsigned short* wmat, unsigned short* bmat)
{
    const int Lout = Lin * 2;
    const long long N = 64LL * Lout;
    // tconv: k=4, stride 2, p=1 -> effective pad 2; lhs dilation handled in im2col
    run_conv(s, src, sC, sB, Lin, cin, wa, ba, cout, 4, 2, 2, 1, nullptr, 1,
             H1, N, Lout, Lout, wmat, bmat);
    run_conv(s, H1, N, Lout, Lout, cout, wb, bb, cout, 3, 1, 1, 0, nullptr, 0,
             H2, N, Lout, Lout, wmat, bmat);
    run_conv(s, src, sC, sB, Lin, cin, wsk, bsk, cout, 4, 2, 2, 1, H2, 1,
             out, osM, osB, Lout, wmat, bmat);
}

extern "C" void kernel_launch(void* const* d_in, const int* in_sizes, int n_in,
                              void* d_out, int out_size, void* d_ws, size_t ws_size,
                              hipStream_t stream)
{
    (void)in_sizes; (void)n_in; (void)out_size; (void)ws_size;

    const int B = 64;
    const int L0 = 8192, L1 = 4096, L2 = 2048, L3 = 1024;
    const int C0 = 8, C1 = 128, C3 = 512;
    const int N1 = B * L1, N2 = B * L2, N3 = B * L3;

    // ---- inputs (setup_inputs() dict insertion order) ----
    // 0: x ; enc blocks i: [1+6i..6+6i] = wa,ba,wb,bb,ws,bs ; dec at 19+6i ; 37: codebook
    const float* X  = (const float*)d_in[0];
    const float* CB = (const float*)d_in[37];
    auto EP = [&](int blk, int j) { return (const float*)d_in[1 + 6 * blk + j]; };
    auto DP = [&](int blk, int j) { return (const float*)d_in[19 + 6 * blk + j]; };

    // ---- workspace carve-up ----
    char* ws = (char*)d_ws;
    size_t off = 0;
    auto alloc = [&](size_t bytes) -> void* {
        void* p = ws + off;
        off = (off + bytes + 255) & ~(size_t)255;
        return p;
    };
    const size_t ACT = (size_t)C1 * N1 * 4;       // 33.5M floats, max activation size
    float* R0 = (float*)alloc(ACT);
    float* R1 = (float*)alloc(ACT);
    float* R2 = (float*)alloc(ACT);
    float* H1 = (float*)alloc(ACT);
    float* H2 = (float*)alloc(ACT);
    unsigned short* BMAT = (unsigned short*)alloc((size_t)2048 * N2 * 2); // max N*Kp
    unsigned short* WMAT = (unsigned short*)alloc((size_t)512 * 2048 * 2);
    float* SCORES = (float*)alloc((size_t)256 * N3 * 4);
    float* CNORM  = (float*)alloc(256 * 4);
    int*   IDX    = (int*)alloc((size_t)N3 * 4);
    int*   HIST   = (int*)alloc(256 * 4);
    float* ACCUM  = (float*)alloc(64);

    // ---- encoder (activations channel-major: buf[c*N + b*L + l]) ----
    down_block(stream, X, L0, (long long)C0 * L0, L0, 8, 128,
               EP(0,0), EP(0,1), EP(0,2), EP(0,3), EP(0,4), EP(0,5),
               R0, N1, L1, H1, H2, WMAT, BMAT);
    down_block(stream, R0, N1, L1, L1, 128, 256,
               EP(1,0), EP(1,1), EP(1,2), EP(1,3), EP(1,4), EP(1,5),
               R1, N2, L2, H1, H2, WMAT, BMAT);
    down_block(stream, R1, N2, L2, L2, 256, 512,
               EP(2,0), EP(2,1), EP(2,2), EP(2,3), EP(2,4), EP(2,5),
               R2, N3, L3, H1, H2, WMAT, BMAT);

    // ---- VQ: scores = codebook x Z (WMMA GEMM), then argmin / quantize ----
    vqvae_pack_weight<<<512, 256, 0, stream>>>(CB, 256, 512, 1, 256, 512, 0, WMAT);
    {
        long long total = (long long)512 * N3;
        vqvae_im2col<<<(int)((total + 255) / 256), 256, 0, stream>>>(
            R2, N3, L3, L3, 512, 1, 1, 0, L3, N3, 512, 0, BMAT);
    }
    {
        dim3 grid(N3 / 128, 4);
        vqvae_gemm_wmma_bf16<<<grid, 256, 0, stream>>>(WMAT, BMAT, nullptr, nullptr,
                                                       SCORES, 256, 256, N3, 512,
                                                       L3, N3, L3, 0);
    }
    vqvae_cnorm<<<1, 256, 0, stream>>>(CB, CNORM, 512);
    vqvae_reset<<<1, 256, 0, stream>>>(HIST, ACCUM);
    vqvae_argmin<<<N3 / 256, 256, 0, stream>>>(SCORES, CNORM, IDX, HIST, N3);
    {
        long long total = (long long)C3 * N3;
        long long nb = (total + 255) / 256;
        int blocks = (int)(nb > 1048576 ? 1048576 : nb);
        vqvae_quant_loss<<<blocks, 256, 0, stream>>>(R2, CB, IDX, R0, ACCUM, N3, 512);
    }

    // ---- decoder (input q in R0, channel-major) ----
    up_block(stream, R0, N3, L3, L3, 512, 256,
             DP(0,0), DP(0,1), DP(0,2), DP(0,3), DP(0,4), DP(0,5),
             R1, N2, L2, H1, H2, WMAT, BMAT);
    up_block(stream, R1, N2, L2, L2, 256, 128,
             DP(1,0), DP(1,1), DP(1,2), DP(1,3), DP(1,4), DP(1,5),
             R2, N1, L1, H1, H2, WMAT, BMAT);
    // final block writes (B, C=8, L=8192) straight into d_out
    up_block(stream, R2, N1, L1, L1, 128, 8,
             DP(2,0), DP(2,1), DP(2,2), DP(2,3), DP(2,4), DP(2,5),
             (float*)d_out, L0, (long long)C0 * L0, H1, H2, WMAT, BMAT);

    // ---- scalars: loss, perplexity appended after the decoder output ----
    vqvae_finalize<<<1, 32, 0, stream>>>(HIST, ACCUM,
                                         (float*)d_out + (size_t)B * C0 * L0,
                                         (float)N3, (float)((size_t)C3 * N3));
}
